// MultiheadAttention_90383291777199
// MI455X (gfx1250) — compile-verified, matrix operand order checked by reading, not run on hardware
//
#include <hip/hip_runtime.h>

// ---------------- CDNA5 (gfx1250) wave32 WMMA MHA ----------------
// L=2048, B=2, E=1024, H=16, HD=64.  bf16 WMMA, f32 accumulate,
// flash-attention (no materialized scores). One-shot bf16 conversion so
// GEMM inner loops are pure load_b128 + v_wmma.

typedef __attribute__((ext_vector_type(16))) __bf16 v16bf;
typedef __attribute__((ext_vector_type(8)))  __bf16 v8bf;
typedef __attribute__((ext_vector_type(4)))  __bf16 v4bf;
typedef __attribute__((ext_vector_type(8)))  float  v8f;
typedef __attribute__((ext_vector_type(4)))  float  v4f;

#define L_SEQ 2048
#define BATCH 2
#define EMB   1024
#define NHEAD 16
#define HD    64
#define NTOK  (L_SEQ * BATCH)   // 4096 tokens
#define NH    (BATCH * NHEAD)   // 32 batched heads

union BfTile { v8bf h[2]; v16bf v; };

__device__ __forceinline__ v8f wmma_bf16(v16bf a, v16bf b, v8f c) {
  // v_wmma_f32_16x16x32_bf16: D = A(16x32) * B(32x16) + C(16x16 f32)
  return __builtin_amdgcn_wmma_f32_16x16x32_bf16(
      false, a, false, b, (short)0, c, false, false);
}

// 16x32 bf16 tile in ISA A/B striping: lane%16 -> row (A) / column (B);
// lane/16 picks K-halves {0..7,16..23} vs {8..15,24..31}.
__device__ __forceinline__ v16bf load_tile(const __bf16* p, int ld, int lane) {
  const int rc = lane & 15, hp = lane >> 4;
  const __bf16* r = p + rc * ld + hp * 8;
  BfTile t;
  t.h[0] = *(const v8bf*)(r);
  t.h[1] = *(const v8bf*)(r + 16);
  return t.v;
}

// ---------------- Kernel 0: one-shot fp32 -> bf16 ----------------
__global__ void __launch_bounds__(256)
cvt_bf16_kernel(const float* __restrict__ src, __bf16* __restrict__ dst, int n4) {
  const int stride = gridDim.x * blockDim.x;
  for (int i = blockIdx.x * blockDim.x + threadIdx.x; i < n4; i += stride) {
    v4f f = *(const v4f*)(src + (size_t)i * 4);
    *(v4bf*)(dst + (size_t)i * 4) = __builtin_convertvector(f, v4bf);
  }
}

// ---------------- Kernel 1: fused in-projection ----------------
__global__ void __launch_bounds__(256)
mha_inproj_kernel(const __bf16* __restrict__ Xq, const __bf16* __restrict__ Xk,
                  const __bf16* __restrict__ Xv, const __bf16* __restrict__ Wi,
                  const float* __restrict__ bias,
                  __bf16* __restrict__ Qb, __bf16* __restrict__ Kb,
                  __bf16* __restrict__ VTb) {
  const int lane = threadIdx.x & 31;
  const int ws   = (blockIdx.x * blockDim.x + threadIdx.x) >> 5;
  const int mblk   = ws & 127;          // 128 token blocks of 32
  const int nstrip = ws >> 7;           // 48 strips of 64 features (3*E)
  const int f0   = nstrip * 64;
  const int proj = f0 >> 10;            // 0=q, 1=k, 2=v
  const __bf16* X = (proj == 0) ? Xq : ((proj == 1) ? Xk : Xv);
  const int t0 = mblk * 32;

  const v8f zero = {0.f, 0.f, 0.f, 0.f, 0.f, 0.f, 0.f, 0.f};
  v8f acc[2][4] = {{zero, zero, zero, zero}, {zero, zero, zero, zero}};

  for (int k0 = 0; k0 < EMB; k0 += 32) {
    const v16bf a0 = load_tile(X + (size_t)t0 * EMB + k0,        EMB, lane);
    const v16bf a1 = load_tile(X + (size_t)(t0 + 16) * EMB + k0, EMB, lane);
#pragma unroll
    for (int nt = 0; nt < 4; ++nt) {
      const v16bf bm = load_tile(Wi + (size_t)(f0 + nt * 16) * EMB + k0, EMB, lane);
      acc[0][nt] = wmma_bf16(a0, bm, acc[0][nt]);
      acc[1][nt] = wmma_bf16(a1, bm, acc[1][nt]);
    }
  }

  const float scale = (proj == 0) ? 0.125f : 1.0f;   // HD^-0.5 folded into Q
  const int col = lane & 15, hp = lane >> 4;
#pragma unroll
  for (int mt = 0; mt < 2; ++mt)
#pragma unroll
    for (int nt = 0; nt < 4; ++nt) {
      const int f  = f0 + nt * 16 + col;
      const float bval = bias[f];
      const int ff = f & (EMB - 1);
      const int h = ff >> 6, d = ff & 63;
#pragma unroll
      for (int r = 0; r < 8; ++r) {
        const int t = t0 + mt * 16 + r + hp * 8;
        const int l = t >> 1, b = t & 1;          // token = l*B + b
        const int nh = b * NHEAD + h;
        const __bf16 obf = (__bf16)((acc[mt][nt][r] + bval) * scale);
        if (proj == 0)      Qb[((size_t)nh * L_SEQ + l) * HD + d] = obf;
        else if (proj == 1) Kb[((size_t)nh * L_SEQ + l) * HD + d] = obf;
        else                VTb[((size_t)nh * HD + d) * L_SEQ + l] = obf;
      }
    }
}

// ---------------- Kernel 2: flash attention ----------------
// One wave per (head, 16-row query block); independent waves, no block barrier.
__global__ void __launch_bounds__(256)
mha_attn_kernel(const __bf16* __restrict__ Qb, const __bf16* __restrict__ Kb,
                const __bf16* __restrict__ VTb, __bf16* __restrict__ Ob) {
  __shared__ alignas(16) __bf16 stile[8][16][40];  // per-wave P staging
  const int lane = threadIdx.x & 31;
  const int wid  = threadIdx.x >> 5;
  const int ws   = (blockIdx.x * blockDim.x + threadIdx.x) >> 5;
  const int nh   = ws >> 7;                     // 32 heads
  const int mblk = ws & 127;                    // 128 query blocks of 16
  const int m0   = mblk * 16;
  const int col = lane & 15, hp = lane >> 4;
  const float LOG2E = 1.44269504f;

  const __bf16* Qh = Qb + (size_t)nh * L_SEQ * HD;
  const __bf16* Kh = Kb + (size_t)nh * L_SEQ * HD;
  const __bf16* Vh = VTb + (size_t)nh * HD * L_SEQ;

  const v16bf aq0 = load_tile(Qh + m0 * HD + 0,  HD, lane);
  const v16bf aq1 = load_tile(Qh + m0 * HD + 32, HD, lane);

  const v8f zero = {0.f, 0.f, 0.f, 0.f, 0.f, 0.f, 0.f, 0.f};
  v8f acc[4] = {zero, zero, zero, zero};        // O tile: 16 rows x 64 dims
  float mrow[8], lrow[8];
#pragma unroll
  for (int r = 0; r < 8; ++r) { mrow[r] = -1.0e30f; lrow[r] = 0.f; }

  for (int j0 = 0; j0 < L_SEQ; j0 += 32) {
    const v16bf b00 = load_tile(Kh + (size_t)j0 * HD + 0,         HD, lane);
    const v16bf b01 = load_tile(Kh + (size_t)j0 * HD + 32,        HD, lane);
    const v16bf b10 = load_tile(Kh + (size_t)(j0 + 16) * HD + 0,  HD, lane);
    const v16bf b11 = load_tile(Kh + (size_t)(j0 + 16) * HD + 32, HD, lane);
    v8f s0 = zero, s1 = zero;
    s0 = wmma_bf16(aq0, b00, s0);
    s0 = wmma_bf16(aq1, b01, s0);
    s1 = wmma_bf16(aq0, b10, s1);
    s1 = wmma_bf16(aq1, b11, s1);

    // Online softmax. Row (r + 8*hp) lives in one VGPR across 16 lanes.
    float alpha[8];
#pragma unroll
    for (int r = 0; r < 8; ++r) {
      float mx = fmaxf(s0[r], s1[r]);
#pragma unroll
      for (int off = 8; off >= 1; off >>= 1)
        mx = fmaxf(mx, __shfl_xor(mx, off, 16));
      const float mnew = fmaxf(mrow[r], mx);
      const float p0 = __builtin_amdgcn_exp2f((s0[r] - mnew) * LOG2E);
      const float p1 = __builtin_amdgcn_exp2f((s1[r] - mnew) * LOG2E);
      float ps = p0 + p1;
#pragma unroll
      for (int off = 8; off >= 1; off >>= 1)
        ps += __shfl_xor(ps, off, 16);
      alpha[r] = __builtin_amdgcn_exp2f((mrow[r] - mnew) * LOG2E);
      lrow[r]  = lrow[r] * alpha[r] + ps;
      mrow[r]  = mnew;
      stile[wid][r + hp * 8][col]      = (__bf16)p0;
      stile[wid][r + hp * 8][16 + col] = (__bf16)p1;
    }
#pragma unroll
    for (int nt = 0; nt < 4; ++nt)
#pragma unroll
      for (int r = 0; r < 8; ++r) acc[nt][r] *= alpha[r];

    // DS ops are in-order within a wave: drain stores, then reread in A layout.
    asm volatile("s_wait_dscnt 0" ::: "memory");
    BfTile pa;
    {
      const __bf16* pr = &stile[wid][col][hp * 8];
      pa.h[0] = *(const v8bf*)(pr);
      pa.h[1] = *(const v8bf*)(pr + 16);
    }

    // O += P(16x32) @ V(32x64): B columns come from transposed V rows.
#pragma unroll
    for (int nt = 0; nt < 4; ++nt) {
      const v16bf bv = load_tile(Vh + (size_t)(nt * 16) * L_SEQ + j0, L_SEQ, lane);
      acc[nt] = wmma_bf16(pa.v, bv, acc[nt]);
    }
  }

#pragma unroll
  for (int r = 0; r < 8; ++r) {
    const float inv = __builtin_amdgcn_rcpf(lrow[r]);
    const int row = r + hp * 8;
#pragma unroll
    for (int nt = 0; nt < 4; ++nt)
      Ob[((size_t)nh * L_SEQ + (m0 + row)) * HD + nt * 16 + col] =
          (__bf16)(acc[nt][r] * inv);
  }
}

// ---------------- Kernel 3: output projection ----------------
__global__ void __launch_bounds__(256)
mha_outproj_kernel(const __bf16* __restrict__ Ob, const __bf16* __restrict__ Wo,
                   const float* __restrict__ bout, float* __restrict__ out) {
  const int lane = threadIdx.x & 31;
  const int ws   = (blockIdx.x * blockDim.x + threadIdx.x) >> 5;
  const int mblk   = ws & 127;          // 128 token blocks of 32
  const int nstrip = ws >> 7;           // 16 strips of 64 out-features
  const int f0 = nstrip * 64;
  const int t0 = mblk * 32;
  const int col = lane & 15, hp = lane >> 4;

  const v8f zero = {0.f, 0.f, 0.f, 0.f, 0.f, 0.f, 0.f, 0.f};
  v8f acc[2][4] = {{zero, zero, zero, zero}, {zero, zero, zero, zero}};

  for (int k0 = 0; k0 < EMB; k0 += 32) {
    // A: gather concat-head rows; a 32-wide k0 block stays inside one head.
    const int h = k0 >> 6, dbase = k0 & 63;
    BfTile a[2];
#pragma unroll
    for (int mt = 0; mt < 2; ++mt) {
      const int t = t0 + mt * 16 + col, l = t >> 1, b = t & 1;
      const __bf16* ap =
          Ob + (((size_t)(b * NHEAD + h) * L_SEQ + l) * HD) + dbase + hp * 8;
      a[mt].h[0] = *(const v8bf*)(ap);
      a[mt].h[1] = *(const v8bf*)(ap + 16);
    }
#pragma unroll
    for (int nt = 0; nt < 4; ++nt) {
      const v16bf bm = load_tile(Wo + (size_t)(f0 + nt * 16) * EMB + k0, EMB, lane);
      acc[0][nt] = wmma_bf16(a[0].v, bm, acc[0][nt]);
      acc[1][nt] = wmma_bf16(a[1].v, bm, acc[1][nt]);
    }
  }

#pragma unroll
  for (int mt = 0; mt < 2; ++mt)
#pragma unroll
    for (int nt = 0; nt < 4; ++nt) {
      const int f = f0 + nt * 16 + col;
      const float bv = bout[f];
#pragma unroll
      for (int r = 0; r < 8; ++r) {
        const int t = t0 + mt * 16 + r + hp * 8;    // token = l*B + b
        out[(size_t)t * EMB + f] = acc[mt][nt][r] + bv;
      }
    }
}

extern "C" void kernel_launch(void* const* d_in, const int* in_sizes, int n_in,
                              void* d_out, int out_size, void* d_ws, size_t ws_size,
                              hipStream_t stream) {
  const float* query = (const float*)d_in[0];
  const float* key   = (const float*)d_in[1];
  const float* value = (const float*)d_in[2];
  const float* w_in  = (const float*)d_in[3];
  const float* b_in  = (const float*)d_in[4];
  const float* w_out = (const float*)d_in[5];
  const float* b_out = (const float*)d_in[6];
  float* out = (float*)d_out;

  // Workspace (bf16), 64 MB total.
  const size_t ACT = (size_t)NTOK * EMB;           // 4194304
  const size_t WI  = (size_t)3 * EMB * EMB;        // 3145728
  const size_t WO  = (size_t)EMB * EMB;            // 1048576
  const size_t HSZ = (size_t)NH * L_SEQ * HD;      // 4194304
  __bf16* Xq  = (__bf16*)d_ws;
  __bf16* Xk  = Xq + ACT;
  __bf16* Xv  = Xk + ACT;
  __bf16* Wi  = Xv + ACT;
  __bf16* Wo  = Wi + WI;
  __bf16* Qb  = Wo + WO;
  __bf16* Kb  = Qb + HSZ;
  __bf16* VTb = Kb + HSZ;
  __bf16* Ob  = VTb + HSZ;

  // One-shot fp32 -> bf16 conversions (packed v_cvt).
  cvt_bf16_kernel<<<(int)(ACT / 1024), 256, 0, stream>>>(query, Xq, (int)(ACT / 4));
  cvt_bf16_kernel<<<(int)(ACT / 1024), 256, 0, stream>>>(key,   Xk, (int)(ACT / 4));
  cvt_bf16_kernel<<<(int)(ACT / 1024), 256, 0, stream>>>(value, Xv, (int)(ACT / 4));
  cvt_bf16_kernel<<<(int)(WI / 1024),  256, 0, stream>>>(w_in,  Wi, (int)(WI / 4));
  cvt_bf16_kernel<<<(int)(WO / 1024),  256, 0, stream>>>(w_out, Wo, (int)(WO / 4));

  // in-proj: 128 token-blocks x 48 feature-strips = 6144 waves / 8 per block
  mha_inproj_kernel<<<768, 256, 0, stream>>>(Xq, Xk, Xv, Wi, b_in, Qb, Kb, VTb);
  // attention: 32 heads x 128 query-blocks = 4096 waves / 8 per block
  mha_attn_kernel<<<512, 256, 0, stream>>>(Qb, Kb, VTb, Ob);
  // out-proj: 128 token-blocks x 16 feature-strips = 2048 waves / 8 per block
  mha_outproj_kernel<<<256, 256, 0, stream>>>(Ob, Wo, b_out, out);
}
